// PositionAwareSAGE_48885317763310
// MI455X (gfx1250) — matrix operand order repaired and checked by reading, not compiled, not run on hardware
//
#include <hip/hip_runtime.h>
#include <hip/hip_bf16.h>

// PositionAwareSAGE fused inference for gfx1250 (MI455X).
// - All four GEMMs on v_wmma_f32_16x16x32_bf16 (bf16 in, fp32 accumulate).
// - Weight staging into LDS via the Tensor Data Mover (tensor_load_to_lds),
//   with a plain-load fallback if the builtin is unavailable.
// - Mean-aggregation scatter in fp32 with global float atomics (L2-resident).

typedef __attribute__((ext_vector_type(16))) __bf16 v16bf;
typedef __attribute__((ext_vector_type(8)))  __bf16 v8bf;
typedef __attribute__((ext_vector_type(8)))  float  v8f;

#define DIMK 128

#if defined(__gfx1250__) && __has_builtin(__builtin_amdgcn_tensor_load_to_lds) && \
    __has_builtin(__builtin_amdgcn_s_wait_tensorcnt) && __has_builtin(__builtin_amdgcn_groupstaticsize)
#define USE_TDM 1
#else
#define USE_TDM 0
#endif

#if USE_TDM
typedef unsigned int u32x4 __attribute__((ext_vector_type(4)));
typedef int          i32x8 __attribute__((ext_vector_type(8)));
typedef int          i32x4 __attribute__((ext_vector_type(4)));

// One TDM DMA: contiguous `nbytes` of bf16 data (2-byte elements) from global
// `gsrc` into LDS at byte offset `lds_off`.  2-D tensor: dim0 = nbytes/2
// elements, dim1 = 1; tile = whole tensor.  D# layout per CDNA5 ISA ch.8.
static __device__ __forceinline__ void tdm_load_to_lds(const void* gsrc,
                                                       unsigned lds_off,
                                                       unsigned nbytes) {
  const unsigned long long ga = (unsigned long long)(uintptr_t)gsrc;
  const unsigned elems = nbytes >> 1;            // bf16 elements, <= 65535
  u32x4 g0;
  g0[0] = 1u;                                    // count=1 valid descriptor
  g0[1] = lds_off;                               // lds_addr (bytes)
  g0[2] = (unsigned)(ga & 0xffffffffu);          // global_addr[31:0]
  g0[3] = (unsigned)((ga >> 32) & 0x01ffffffu)   // global_addr[56:32]
        | (2u << 30);                            // type = 2 ("image")
  i32x8 g1;
  g1[0] = (1 << 16);                             // data_size=1 (2B); wg_mask=0
  g1[1] = (int)((elems & 0xffffu) << 16);        // tensor_dim0[15:0]  @ bits 63:48
  g1[2] = (int)((elems >> 16) & 0xffffu)         // tensor_dim0[31:16] @ bits 79:64
        | (1 << 16);                             // tensor_dim1 = 1    @ bits 111:80
  g1[3] = (int)((elems & 0xffffu) << 16);        // tile_dim0          @ bits 127:112
  g1[4] = 0;                                     // tile_dim1=0, tile_dim2=0 (1-D tile)
  g1[5] = (int)elems;                            // tensor_dim0_stride[31:0]
  g1[6] = 0;                                     // stride high bits / dim1 stride
  g1[7] = 0;
  const i32x4 z4 = {0, 0, 0, 0};                 // groups 2/3 unused (2-D tensor)
  const i32x8 z8 = {0, 0, 0, 0, 0, 0, 0, 0};
  // 6-arg variant (clang-23 / therock headers): (g0, g1, g2, g3, extra, cpol)
  __builtin_amdgcn_tensor_load_to_lds(g0, g1, z4, z4, z8, 0);
}
#endif

// Stage `nbytes` from global into LDS (whole block participates; returns after
// all waves can safely read the staged data).
static __device__ __forceinline__ void stage_weights(const void* gsrc, void* lds_dst,
                                                     unsigned lds_off, unsigned nbytes) {
#if USE_TDM
  if (threadIdx.x < 32) {                        // one DMA per workgroup (wave 0)
    tdm_load_to_lds(gsrc, lds_off, nbytes);
    __builtin_amdgcn_s_wait_tensorcnt(0);
  }
  __syncthreads();
#else
  const uint4* s = (const uint4*)gsrc;
  uint4* d = (uint4*)lds_dst;
  for (unsigned i = threadIdx.x; i < (nbytes >> 4); i += blockDim.x) d[i] = s[i];
  __syncthreads();
#endif
}

// ---- fragment helpers (wave32 WMMA layouts per CDNA5 ISA 7.12.2) ----
// A 16x32 bf16 fragment: lane L holds row (L&15); element e (0..15) maps to
//   K = kbase + 8*(L>>4) + (e<8 ? e : e+8)  -> two contiguous 8-element runs.
static __device__ __forceinline__ v16bf frag_a_bf16(const __bf16* base) {
  const v8bf lo = *(const v8bf*)(base);
  const v8bf hi = *(const v8bf*)(base + 16);
  v16bf r;
#pragma unroll
  for (int i = 0; i < 8; ++i) { r[i] = lo[i]; r[i + 8] = hi[i]; }
  return r;
}

static __device__ __forceinline__ v16bf frag_a_f32(const float* base, float scale) {
  v16bf r;
#pragma unroll
  for (int i = 0; i < 8; ++i) {
    r[i]     = (__bf16)(base[i]      * scale);
    r[i + 8] = (__bf16)(base[i + 16] * scale);
  }
  return r;
}

// B 32x16 bf16 fragment from W^T stored [n][k]: lane L holds column n=(L&15);
// elements K = kbase + (e<8 ? e : e+8), kbase = k0 + 8*(L>>4) -> two contiguous 8-element runs.
static __device__ __forceinline__ v16bf frag_b(const __bf16* wT, int n, int kbase) {
  return frag_a_bf16(&wT[n * DIMK + kbase]);
}

// ---- kernel 1: bf16-transpose the 4 weight matrices -------------------------
__global__ void k_prep(const float* __restrict__ fp_w, const float* __restrict__ lin_l_w,
                       const float* __restrict__ lin_r_w, const float* __restrict__ res_w,
                       __bf16* __restrict__ fpwT, __bf16* __restrict__ wT3) {
  int idx = blockIdx.x * blockDim.x + threadIdx.x;
  if (idx >= 4 * DIMK * DIMK) return;
  int m = idx >> 14;
  int r = idx & 16383;
  int n = r >> 7, k = r & 127;
  const float* W = (m == 0) ? fp_w : (m == 1) ? lin_l_w : (m == 2) ? lin_r_w : res_w;
  __bf16* T = (m == 0) ? fpwT : (wT3 + (size_t)(m - 1) * DIMK * DIMK);
  T[n * DIMK + k] = (__bf16)W[k * DIMK + n];
}

// ---- kernel 2: zero msg_sum and cnt ----------------------------------------
__global__ void k_zero(float* __restrict__ msg, float* __restrict__ cnt, int n128, int n) {
  int total = n128 + n;
  for (int i = blockIdx.x * blockDim.x + threadIdx.x; i < total; i += gridDim.x * blockDim.x) {
    if (i < n128) msg[i] = 0.0f; else cnt[i - n128] = 0.0f;
  }
}

// ---- kernel 3: xf = [x|pos|len] @ fp_w + fp_b  (WMMA, rank-1 epilogue) ------
__global__ __launch_bounds__(256) void k_gemm1(
    const float* __restrict__ x, const int* __restrict__ pos, const int* __restrict__ len,
    const float* __restrict__ fp_w, const float* __restrict__ fp_b,
    const __bf16* __restrict__ fpwT,
    float* __restrict__ xf_f32, __bf16* __restrict__ xf_bf16, int N) {
  extern __shared__ __bf16 smem1[];
  __bf16* s_w = smem1;   // 32 KB
#if USE_TDM
  stage_weights(fpwT, s_w, __builtin_amdgcn_groupstaticsize(), DIMK * DIMK * 2);
#else
  stage_weights(fpwT, s_w, 0, DIMK * DIMK * 2);
#endif

  const int lane = threadIdx.x & 31;
  const int wave = threadIdx.x >> 5;
  const int wstripe = wave >> 2;
  const int wn = wave & 3;
  const int g = lane >> 4;
  const int j = lane & 15;

  const int ntiles = (N + 31) >> 5;
  for (int tile = blockIdx.x; tile < ntiles; tile += gridDim.x) {
    const int m0 = tile * 32 + wstripe * 16;
    int r = m0 + j; if (r >= N) r = N - 1;
    const float* arow = x + (size_t)r * DIMK;

    v8f acc0 = {}; v8f acc1 = {};
#pragma unroll
    for (int ks = 0; ks < 4; ++ks) {
      const int k0 = ks * 32;
      const v16bf a = frag_a_f32(arow + k0 + 8 * g, 1.0f);
      const v16bf b0 = frag_b(s_w, wn * 32 + j,      k0 + 8 * g);
      const v16bf b1 = frag_b(s_w, wn * 32 + 16 + j, k0 + 8 * g);
      acc0 = __builtin_amdgcn_wmma_f32_16x16x32_bf16(false, a, false, b0, (short)0, acc0, false, false);
      acc1 = __builtin_amdgcn_wmma_f32_16x16x32_bf16(false, a, false, b1, (short)0, acc1, false, false);
    }
#pragma unroll
    for (int t = 0; t < 2; ++t) {
      v8f acc = t ? acc1 : acc0;
      const int n = wn * 32 + t * 16 + j;
      const float bn   = fp_b[n];
      const float w128 = fp_w[128 * DIMK + n];
      const float w129 = fp_w[129 * DIMK + n];
#pragma unroll
      for (int v = 0; v < 8; ++v) {
        const int rr = m0 + v + 8 * g;
        if (rr < N) {
          const float pf = (float)pos[rr] * (1.0f / 50.0f);
          const float lf = (float)len[rr] * (1.0f / 500.0f);
          const float val = acc[v] + bn + pf * w128 + lf * w129;
          xf_f32 [(size_t)rr * DIMK + n] = val;
          xf_bf16[(size_t)rr * DIMK + n] = (__bf16)val;
        }
      }
    }
  }
}

// ---- kernel 4: edge scatter (mean-agg numerator + counts) -------------------
__global__ __launch_bounds__(256) void k_scatter(
    const int* __restrict__ src, const int* __restrict__ dst,
    const float* __restrict__ xf, float* __restrict__ msg,
    float* __restrict__ cnt, int E) {
  const int lane = threadIdx.x & 31;
  int wid = (blockIdx.x * blockDim.x + threadIdx.x) >> 5;
  const int nw = (gridDim.x * blockDim.x) >> 5;
  for (int e = wid; e < E; e += nw) {
    const int s = src[e];
    const int d = dst[e];
    const int en = e + nw;
    if (en < E) __builtin_prefetch(xf + (size_t)src[en] * DIMK + lane * 4, 0, 0);
    const float4 v = ((const float4*)(xf + (size_t)s * DIMK))[lane];
    float* p = msg + (size_t)d * DIMK + lane * 4;
    atomicAdd(p + 0, v.x);
    atomicAdd(p + 1, v.y);
    atomicAdd(p + 2, v.z);
    atomicAdd(p + 3, v.w);
    if (lane == 0) atomicAdd(cnt + d, 1.0f);
  }
}

// ---- kernel 5: fused  relu(agg@Wl + b + xf@Wr) + (xf@Wres + br)  -> score ---
__global__ __launch_bounds__(256) void k_final(
    const float* __restrict__ msg, const float* __restrict__ cnt,
    const __bf16* __restrict__ xf_bf16, const __bf16* __restrict__ wT3,
    const float* __restrict__ lin_l_b, const float* __restrict__ res_b,
    const float* __restrict__ sh_w, const float* __restrict__ sh_b,
    const float* __restrict__ alpha, const float* __restrict__ rrs,
    float* __restrict__ out, int N) {
  extern __shared__ __bf16 smem[];
  __bf16* s_wl   = smem;                 // 32 KB
  __bf16* s_wr   = smem + 16384;         // 32 KB
  __bf16* s_wres = smem + 32768;         // 32 KB
  float*  red    = (float*)(smem + 49152);
#if USE_TDM
  stage_weights(wT3, smem, __builtin_amdgcn_groupstaticsize(), 3 * DIMK * DIMK * 2);
#else
  stage_weights(wT3, smem, 0, 3 * DIMK * DIMK * 2);
#endif

  const float a_sig = 1.0f / (1.0f + __expf(-alpha[0]));
  const int lane = threadIdx.x & 31;
  const int wave = threadIdx.x >> 5;
  const int wstripe = wave >> 2;
  const int wn = wave & 3;
  const int g = lane >> 4;
  const int j = lane & 15;

  const int ntiles = (N + 31) >> 5;
  for (int tile = blockIdx.x; tile < ntiles; tile += gridDim.x) {
    if (threadIdx.x < 32) red[threadIdx.x] = 0.0f;
    __syncthreads();

    const int m0 = tile * 32 + wstripe * 16;
    int r = m0 + j; if (r >= N) r = N - 1;
    const float inv = 1.0f / fmaxf(cnt[r], 1.0f);
    const float*  mrow = msg     + (size_t)r * DIMK;
    const __bf16* xrow = xf_bf16 + (size_t)r * DIMK;

    v8f accc0 = {}, accc1 = {};   // agg@lin_l + xf@lin_r (shared accumulator)
    v8f accr0 = {}, accr1 = {};   // xf@res_w
#pragma unroll
    for (int ks = 0; ks < 4; ++ks) {
      const int k0 = ks * 32;
      const v16bf a_agg = frag_a_f32 (mrow + k0 + 8 * g, inv);
      const v16bf a_xf  = frag_a_bf16(xrow + k0 + 8 * g);
      const int kb = k0 + 8 * g;
      const v16bf bl0 = frag_b(s_wl,   wn * 32 + j,      kb);
      const v16bf bl1 = frag_b(s_wl,   wn * 32 + 16 + j, kb);
      const v16bf br0 = frag_b(s_wr,   wn * 32 + j,      kb);
      const v16bf br1 = frag_b(s_wr,   wn * 32 + 16 + j, kb);
      const v16bf bs0 = frag_b(s_wres, wn * 32 + j,      kb);
      const v16bf bs1 = frag_b(s_wres, wn * 32 + 16 + j, kb);
      accc0 = __builtin_amdgcn_wmma_f32_16x16x32_bf16(false, a_agg, false, bl0, (short)0, accc0, false, false);
      accc1 = __builtin_amdgcn_wmma_f32_16x16x32_bf16(false, a_agg, false, bl1, (short)0, accc1, false, false);
      accc0 = __builtin_amdgcn_wmma_f32_16x16x32_bf16(false, a_xf,  false, br0, (short)0, accc0, false, false);
      accc1 = __builtin_amdgcn_wmma_f32_16x16x32_bf16(false, a_xf,  false, br1, (short)0, accc1, false, false);
      accr0 = __builtin_amdgcn_wmma_f32_16x16x32_bf16(false, a_xf,  false, bs0, (short)0, accr0, false, false);
      accr1 = __builtin_amdgcn_wmma_f32_16x16x32_bf16(false, a_xf,  false, bs1, (short)0, accr1, false, false);
    }
#pragma unroll
    for (int t = 0; t < 2; ++t) {
      v8f ac = t ? accc1 : accc0;
      v8f ar = t ? accr1 : accr0;
      const int n = wn * 32 + t * 16 + j;
      const float bl = lin_l_b[n];
      const float br = res_b[n];
      const float sw = sh_w[n];
#pragma unroll
      for (int v = 0; v < 8; ++v) {
        const float h = fmaxf(ac[v] + bl, 0.0f) + ar[v] + br;
        atomicAdd(&red[wstripe * 16 + v + 8 * g], h * sw);  // ds_add_f32
      }
    }
    __syncthreads();
    if (threadIdx.x < 32) {
      const int rr = tile * 32 + threadIdx.x;
      if (rr < N) {
        const float gscore = red[threadIdx.x] + sh_b[0];
        out[rr] = a_sig * rrs[rr] + (1.0f - a_sig) * gscore;
      }
    }
    __syncthreads();
  }
}

// ---- host launch ------------------------------------------------------------
extern "C" void kernel_launch(void* const* d_in, const int* in_sizes, int n_in,
                              void* d_out, int out_size, void* d_ws, size_t ws_size,
                              hipStream_t stream) {
  const float* x       = (const float*)d_in[0];
  const int*   pos     = (const int*)  d_in[1];
  const int*   len     = (const int*)  d_in[2];
  const int*   ei      = (const int*)  d_in[3];
  const float* rrs     = (const float*)d_in[4];
  const float* fp_w    = (const float*)d_in[5];
  const float* fp_b    = (const float*)d_in[6];
  const float* lin_l_w = (const float*)d_in[7];
  const float* lin_l_b = (const float*)d_in[8];
  const float* lin_r_w = (const float*)d_in[9];
  const float* res_w   = (const float*)d_in[10];
  const float* res_b   = (const float*)d_in[11];
  const float* sh_w    = (const float*)d_in[12];
  const float* sh_b    = (const float*)d_in[13];
  const float* alpha   = (const float*)d_in[14];

  const int N = in_sizes[1];
  const int E = in_sizes[3] / 2;
  const int* src = ei;
  const int* dst = ei + E;

  char* ws = (char*)d_ws;
  size_t off = 0;
  auto take = [&](size_t bytes) -> void* {
    void* p = ws + off;
    off += (bytes + 255) & ~(size_t)255;
    return p;
  };
  float*  xf_f32  = (float*) take((size_t)N * DIMK * 4);
  float*  msg     = (float*) take((size_t)N * DIMK * 4);
  __bf16* xf_bf16 = (__bf16*)take((size_t)N * DIMK * 2);
  float*  cnt     = (float*) take((size_t)N * 4);
  __bf16* fpwT    = (__bf16*)take((size_t)DIMK * DIMK * 2);
  __bf16* wT3     = (__bf16*)take((size_t)3 * DIMK * DIMK * 2);
  (void)ws_size; (void)n_in; (void)out_size;

  const int row_tiles = (N + 31) / 32;

  k_prep<<<(4 * DIMK * DIMK + 255) / 256, 256, 0, stream>>>(
      fp_w, lin_l_w, lin_r_w, res_w, fpwT, wT3);

  k_zero<<<(N * (DIMK + 1) + 255) / 256, 256, 0, stream>>>(msg, cnt, N * DIMK, N);

  k_gemm1<<<row_tiles, 256, (size_t)DIMK * DIMK * 2, stream>>>(
      x, pos, len, fp_w, fp_b, fpwT, xf_f32, xf_bf16, N);

  k_scatter<<<(E + 7) / 8, 256, 0, stream>>>(src, dst, xf_f32, msg, cnt, E);

  const size_t smem_bytes = 3 * DIMK * DIMK * sizeof(__bf16) + 32 * sizeof(float) + 256;
  k_final<<<row_tiles, 256, smem_bytes, stream>>>(
      msg, cnt, xf_bf16, wT3, lin_l_b, res_b, sh_w, sh_b, alpha, rrs,
      (float*)d_out, N);
}
